// SpikeOneStepDecoder_68255620268198
// MI455X (gfx1250) — matrix operand order repaired and checked by reading, not compiled, not run on hardware
//
#include <hip/hip_runtime.h>
#include <stddef.h>

// ---------- problem constants ----------
#define V_SZ 50257
#define E_SZ 512
#define H_SZ 1024
#define B_SZ 1024
#define TSTEPS 10

// ---------- vector types ----------
typedef __attribute__((ext_vector_type(16))) __bf16         v16bf;
typedef __attribute__((ext_vector_type(8)))  float          v8f;
typedef __attribute__((ext_vector_type(8)))  unsigned short ushort8;
typedef __attribute__((ext_vector_type(4)))  unsigned int   uint4v;
typedef __attribute__((ext_vector_type(2)))  unsigned int   uint2v;
typedef __attribute__((ext_vector_type(4)))  float          float4v;

// round-to-nearest-even f32 -> bf16 bits
__device__ __forceinline__ unsigned short f32_to_bf16_bits(float f) {
  union { float f; unsigned int u; } v; v.f = f;
  unsigned int r = v.u + 0x7FFFu + ((v.u >> 16) & 1u);
  return (unsigned short)(r >> 16);
}
__device__ __forceinline__ unsigned int pack_bf16_pair(float a, float b) {
  return (unsigned int)f32_to_bf16_bits(a) |
         ((unsigned int)f32_to_bf16_bits(b) << 16);
}

// ---------- gfx1250 async copy: global -> LDS, 16 bytes/lane (ASYNCcnt) ----------
__device__ __forceinline__ void async_ld16(unsigned lds_off, const void* gptr) {
  asm volatile("global_load_async_to_lds_b128 %0, %1, off"
               :: "v"(lds_off), "v"(gptr) : "memory");
}
__device__ __forceinline__ void wait_async0() {
  asm volatile("s_wait_asynccnt 0x0" ::: "memory");
}

// ---------- small elementwise kernels ----------
__global__ void embed_mask_bf16_kernel(const int* __restrict__ tok,
                                       const float* __restrict__ emb,
                                       const float* __restrict__ mask,
                                       unsigned short* __restrict__ x_bf) {
  int i = blockIdx.x * blockDim.x + threadIdx.x;   // B*E
  int b = i >> 9;
  int e = i & (E_SZ - 1);
  x_bf[i] = f32_to_bf16_bits(emb[(size_t)tok[b] * E_SZ + e] * mask[i]);
}

__global__ void embed_mask_f32_kernel(const int* __restrict__ tok,
                                      const float* __restrict__ emb,
                                      const float* __restrict__ mask,
                                      float* __restrict__ x) {
  int i = blockIdx.x * blockDim.x + threadIdx.x;   // B*E
  int b = i >> 9;
  int e = i & (E_SZ - 1);
  x[i] = emb[(size_t)tok[b] * E_SZ + e] * mask[i];
}

__global__ void init_state_kernel(const float* __restrict__ hidden,
                                  float* __restrict__ mem_lif,
                                  float* __restrict__ mem_int) {
  int i = blockIdx.x * blockDim.x + threadIdx.x;   // B*H
  mem_lif[i] = hidden[i];
  mem_int[i] = 0.0f;
}

__global__ void lif_step_bf16_kernel(const float* __restrict__ delta1,
                                     const float* __restrict__ mask_lif,
                                     float* __restrict__ mem_lif,
                                     unsigned short* __restrict__ h_bf,
                                     const float* __restrict__ thr_p,
                                     const float* __restrict__ leak_p) {
  int i = blockIdx.x * blockDim.x + threadIdx.x;   // B*H
  const float thr  = thr_p[0];
  const float leak = leak_p[0];
  float mem   = leak * mem_lif[i] + delta1[i];
  float spike = ((mem / thr - 1.0f) > 0.0f) ? 1.0f : 0.0f;
  mem_lif[i]  = mem - thr * spike;
  h_bf[i]     = f32_to_bf16_bits(spike * mask_lif[i]);
}

__global__ void lif_step_f32_kernel(const float* __restrict__ delta1,
                                    const float* __restrict__ mask_lif,
                                    float* __restrict__ mem_lif,
                                    float* __restrict__ h,
                                    const float* __restrict__ thr_p,
                                    const float* __restrict__ leak_p) {
  int i = blockIdx.x * blockDim.x + threadIdx.x;   // B*H
  const float thr  = thr_p[0];
  const float leak = leak_p[0];
  float mem   = leak * mem_lif[i] + delta1[i];
  float spike = ((mem / thr - 1.0f) > 0.0f) ? 1.0f : 0.0f;
  mem_lif[i]  = mem - thr * spike;
  h[i]        = spike * mask_lif[i];
}

// bulk f32 -> bf16 conversion, 4 elements/thread (n must be multiple of 4)
__global__ void cvt_f32_bf16_kernel(const float* __restrict__ src,
                                    unsigned short* __restrict__ dst) {
  size_t i = ((size_t)blockIdx.x * blockDim.x + threadIdx.x) * 4;
  float4v f = *(const float4v*)(src + i);
  uint2v w;
  w[0] = pack_bf16_pair(f[0], f[1]);
  w[1] = pack_bf16_pair(f[2], f[3]);
  *(uint2v*)(dst + i) = w;
}

// ---------- GEMM tiling ----------
#define BM 128
#define BN 128
#define BK 32

// ===================================================================
// Fast path: A,B already bf16 in global; LDS tiles filled by async DMA
// C[M,N] = A[M,K] * B[N,K]^T (+bias) (+=C)
// ===================================================================
__global__ __launch_bounds__(256)
void gemm_async_bf16_wmma_kernel(const unsigned short* __restrict__ Abf,
                                 const unsigned short* __restrict__ Bbf,
                                 const float* __restrict__ bias,
                                 float* __restrict__ C,
                                 int M, int N, int K, int accumulate) {
  __shared__ __align__(16) unsigned short As[2 * BM * BK];
  __shared__ __align__(16) unsigned short Bs[2 * BN * BK];

  const int tid  = threadIdx.x;
  const int lane = tid & 31;
  const int wave = tid >> 5;       // 0..7
  const int wrow = wave >> 2;      // m offset wrow*64
  const int wcol = wave & 3;       // n offset wcol*32
  const int blockM = blockIdx.y * BM;
  const int blockN = blockIdx.x * BN;

  v8f acc[4][2];
#pragma unroll
  for (int i = 0; i < 4; ++i)
#pragma unroll
    for (int j = 0; j < 2; ++j) acc[i][j] = (v8f)0.0f;

  // 16-byte chunks: tile = BM*BK bf16 = 512 chunks; thread owns {tid, tid+256}
  const unsigned as_base = (unsigned)(size_t)(void*)As;
  const unsigned bs_base = (unsigned)(size_t)(void*)Bs;
  const unsigned short* ag[2];
  const unsigned short* bg[2];
  unsigned asofs[2], bsofs[2];
  int  crow[2], coff[2];
  bool bval[2];
#pragma unroll
  for (int q = 0; q < 2; ++q) {
    int c   = tid + q * 256;
    int row = c >> 2;
    int off = (c & 3) * 8;
    crow[q] = row; coff[q] = off;
    ag[q]   = Abf + (size_t)(blockM + row) * K + off;
    int n   = blockN + row;
    bval[q] = (n < N);
    bg[q]   = Bbf + (size_t)n * K + off;
    asofs[q] = as_base + (unsigned)((row * BK + off) * 2);
    bsofs[q] = bs_base + (unsigned)((row * BK + off) * 2);
  }

  // zero-fill OOB B rows in both buffers (never async-written)
#pragma unroll
  for (int q = 0; q < 2; ++q) {
    if (!bval[q]) {
      uint4v z = (uint4v)0u;
      *(uint4v*)&Bs[0 * BN * BK + crow[q] * BK + coff[q]] = z;
      *(uint4v*)&Bs[1 * BN * BK + crow[q] * BK + coff[q]] = z;
    }
  }

  // prologue: DMA tile 0 into buffer 0
#pragma unroll
  for (int q = 0; q < 2; ++q) {
    async_ld16(asofs[q], ag[q]);
    if (bval[q]) async_ld16(bsofs[q], bg[q]);
  }

  const int mrow = lane & 15;
  const int lk8  = (lane >> 4) * 8;
  const int lk16 = (lane >> 4) * 16;

  const int nK = K / BK;
  for (int kt = 0; kt < nK; ++kt) {
    const int buf = kt & 1;
    wait_async0();       // DMA for current buffer complete (this wave's issues)
    __syncthreads();     // all waves' DMA complete; prev buffer free

    // issue DMA for next tile into the other buffer; hidden under WMMAs
    if (kt + 1 < nK) {
      const unsigned bofs = (unsigned)((buf ^ 1) * (BM * BK * 2));
      const size_t   kofs = (size_t)(kt + 1) * BK;
#pragma unroll
      for (int q = 0; q < 2; ++q) {
        async_ld16(asofs[q] + bofs, ag[q] + kofs);
        if (bval[q]) async_ld16(bsofs[q] + bofs, bg[q] + kofs);
      }
    }

    const unsigned short* Asb = As + buf * (BM * BK);
    const unsigned short* Bsb = Bs + buf * (BN * BK);

    union FU { v16bf v; ushort8 h[2]; };
    FU a[4], b[2];
#pragma unroll
    for (int i = 0; i < 4; ++i) {
      int r = wrow * 64 + i * 16 + mrow;
      a[i].h[0] = *(const ushort8*)&Asb[r * BK + lk8];
      a[i].h[1] = *(const ushort8*)&Asb[r * BK + 16 + lk8];
    }
#pragma unroll
    for (int j = 0; j < 2; ++j) {
      int n = wcol * 32 + j * 16 + mrow;
      b[j].h[0] = *(const ushort8*)&Bsb[n * BK + lk8];
      b[j].h[1] = *(const ushort8*)&Bsb[n * BK + 16 + lk8];
    }
#pragma unroll
    for (int i = 0; i < 4; ++i)
#pragma unroll
      for (int j = 0; j < 2; ++j)
        acc[i][j] = __builtin_amdgcn_wmma_f32_16x16x32_bf16(
            false, a[i].v, false, b[j].v, (short)0, acc[i][j], false, false);
  }

  const int crow0 = blockM + wrow * 64 + ((lane >> 4) * 8);
  const int ccol0 = blockN + wcol * 32 + (lane & 15);
#pragma unroll
  for (int j = 0; j < 2; ++j) {
    int col = ccol0 + j * 16;
    if (col < N) {
      float bval2 = bias ? bias[col] : 0.0f;
#pragma unroll
      for (int i = 0; i < 4; ++i) {
#pragma unroll
        for (int r = 0; r < 8; ++r) {
          int row = crow0 + i * 16 + r;
          size_t idx = (size_t)row * (size_t)N + (size_t)col;
          float v = acc[i][j][r] + bval2;
          if (accumulate) v += C[idx];
          C[idx] = v;
        }
      }
    }
  }
}

// ===================================================================
// Fallback path (small ws): f32 sources, on-the-fly bf16 conversion
// (round-2 kernel)
// ===================================================================
__device__ __forceinline__ void load16(const float* __restrict__ gp,
                                       float4v fr[4], bool valid) {
  if (valid) {
#pragma unroll
    for (int v = 0; v < 4; ++v) fr[v] = *(const float4v*)(gp + 4 * v);
  } else {
#pragma unroll
    for (int v = 0; v < 4; ++v) fr[v] = (float4v)0.0f;
  }
}
__device__ __forceinline__ void cvt_store16(const float4v fr[4],
                                            unsigned short* lp) {
  uint4v lo, hi;
#pragma unroll
  for (int i = 0; i < 4; ++i) {
    lo[i] = pack_bf16_pair(fr[i >> 1][(i & 1) * 2],       fr[i >> 1][(i & 1) * 2 + 1]);
    hi[i] = pack_bf16_pair(fr[2 + (i >> 1)][(i & 1) * 2], fr[2 + (i >> 1)][(i & 1) * 2 + 1]);
  }
  *(uint4v*)(lp)     = lo;
  *(uint4v*)(lp + 8) = hi;
}

__global__ __launch_bounds__(256)
void gemm_bf16_wmma_kernel(const float* __restrict__ A,
                           const float* __restrict__ Bw,
                           const float* __restrict__ bias,
                           float* __restrict__ C,
                           int M, int N, int K, int accumulate) {
  __shared__ __align__(16) unsigned short As[2 * BM * BK];
  __shared__ __align__(16) unsigned short Bs[2 * BN * BK];

  const int tid  = threadIdx.x;
  const int lane = tid & 31;
  const int wave = tid >> 5;
  const int wrow = wave >> 2;
  const int wcol = wave & 3;
  const int blockM = blockIdx.y * BM;
  const int blockN = blockIdx.x * BN;

  v8f acc[4][2];
#pragma unroll
  for (int i = 0; i < 4; ++i)
#pragma unroll
    for (int j = 0; j < 2; ++j) acc[i][j] = (v8f)0.0f;

  const int lr = tid >> 1;
  const int lc = (tid & 1) * 16;
  const int mrow = lane & 15;
  const int lk8  = (lane >> 4) * 8;
  const int lk16 = (lane >> 4) * 16;

  const int    bn     = blockN + lr;
  const float* arow   = A  + (size_t)(blockM + lr) * K + lc;
  const float* brow   = Bw + (size_t)bn * K + lc;
  const bool   bvalid = (bn < N);

  float4v fa[4], fb[4];
  load16(arow, fa, true);
  load16(brow, fb, bvalid);

  const int nK = K / BK;
  for (int kt = 0; kt < nK; ++kt) {
    const int buf = kt & 1;
    unsigned short* Asb = As + buf * (BM * BK);
    unsigned short* Bsb = Bs + buf * (BN * BK);

    cvt_store16(fa, &Asb[lr * BK + lc]);
    cvt_store16(fb, &Bsb[lr * BK + lc]);
    __syncthreads();

    if (kt + 1 < nK) {
      load16(arow + (size_t)(kt + 1) * BK, fa, true);
      load16(brow + (size_t)(kt + 1) * BK, fb, bvalid);
    }

    union FU { v16bf v; ushort8 h[2]; };
    FU a[4], b[2];
#pragma unroll
    for (int i = 0; i < 4; ++i) {
      int r = wrow * 64 + i * 16 + mrow;
      a[i].h[0] = *(const ushort8*)&Asb[r * BK + lk8];
      a[i].h[1] = *(const ushort8*)&Asb[r * BK + 16 + lk8];
    }
#pragma unroll
    for (int j = 0; j < 2; ++j) {
      int n = wcol * 32 + j * 16 + mrow;
      b[j].h[0] = *(const ushort8*)&Bsb[n * BK + lk8];
      b[j].h[1] = *(const ushort8*)&Bsb[n * BK + 16 + lk8];
    }
#pragma unroll
    for (int i = 0; i < 4; ++i)
#pragma unroll
      for (int j = 0; j < 2; ++j)
        acc[i][j] = __builtin_amdgcn_wmma_f32_16x16x32_bf16(
            false, a[i].v, false, b[j].v, (short)0, acc[i][j], false, false);
  }

  const int crow0 = blockM + wrow * 64 + ((lane >> 4) * 8);
  const int ccol0 = blockN + wcol * 32 + (lane & 15);
#pragma unroll
  for (int j = 0; j < 2; ++j) {
    int col = ccol0 + j * 16;
    if (col < N) {
      float bval = bias ? bias[col] : 0.0f;
#pragma unroll
      for (int i = 0; i < 4; ++i) {
#pragma unroll
        for (int r = 0; r < 8; ++r) {
          int row = crow0 + i * 16 + r;
          size_t idx = (size_t)row * (size_t)N + (size_t)col;
          float v = acc[i][j][r] + bval;
          if (accumulate) v += C[idx];
          C[idx] = v;
        }
      }
    }
  }
}

// ---------- host-side launch ----------
extern "C" void kernel_launch(void* const* d_in, const int* in_sizes, int n_in,
                              void* d_out, int out_size, void* d_ws, size_t ws_size,
                              hipStream_t stream) {
  (void)in_sizes; (void)n_in; (void)out_size;

  const int*   tok      = (const int*)  d_in[0];
  const float* hidden   = (const float*)d_in[1];
  const float* emb      = (const float*)d_in[2];
  const float* W1       = (const float*)d_in[3];
  const float* b1       = (const float*)d_in[4];
  const float* W2       = (const float*)d_in[5];
  const float* b2       = (const float*)d_in[6];
  const float* fcW      = (const float*)d_in[7];
  const float* fcb      = (const float*)d_in[8];
  const float* thr      = (const float*)d_in[9];
  const float* leak     = (const float*)d_in[10];
  const float* mask_emb = (const float*)d_in[11];
  const float* mask_lif = (const float*)d_in[12];

  float* linear  = (float*)d_out;                      // [B,V]
  float* mem_int = linear + (size_t)B_SZ * V_SZ;       // [B,H] = 2nd output

  const size_t nBE = (size_t)B_SZ * E_SZ;
  const size_t nBH = (size_t)B_SZ * H_SZ;
  const size_t nHE = (size_t)H_SZ * E_SZ;
  const size_t nHH = (size_t)H_SZ * H_SZ;
  const size_t nVH = (size_t)V_SZ * H_SZ;

  const size_t need_full = 2 * (nBE + nHE + nHH + nBH + nBH + nVH) + 4 * (nBH + nBH);

  if (ws_size >= need_full) {
    // ---- fast path: pre-convert everything to bf16, async-DMA GEMMs ----
    char* p = (char*)d_ws;
    unsigned short* x_bf  = (unsigned short*)p; p += 2 * nBE;
    unsigned short* W1bf  = (unsigned short*)p; p += 2 * nHE;
    unsigned short* W2bf  = (unsigned short*)p; p += 2 * nHH;
    unsigned short* h_bf  = (unsigned short*)p; p += 2 * nBH;
    unsigned short* mi_bf = (unsigned short*)p; p += 2 * nBH;
    unsigned short* fcWbf = (unsigned short*)p; p += 2 * nVH;
    float* delta1  = (float*)p; p += 4 * nBH;
    float* mem_lif = (float*)p;

    embed_mask_bf16_kernel<<<(int)(nBE / 256), 256, 0, stream>>>(tok, emb, mask_emb, x_bf);
    init_state_kernel<<<(int)(nBH / 256), 256, 0, stream>>>(hidden, mem_lif, mem_int);

    cvt_f32_bf16_kernel<<<(int)(nHE / 1024), 256, 0, stream>>>(W1,  W1bf);
    cvt_f32_bf16_kernel<<<(int)(nHH / 1024), 256, 0, stream>>>(W2,  W2bf);
    cvt_f32_bf16_kernel<<<(int)(nVH / 1024), 256, 0, stream>>>(fcW, fcWbf);

    {
      dim3 grid(H_SZ / BN, B_SZ / BM);
      gemm_async_bf16_wmma_kernel<<<grid, 256, 0, stream>>>(
          x_bf, W1bf, b1, delta1, B_SZ, H_SZ, E_SZ, 0);
    }
    for (int t = 0; t < TSTEPS; ++t) {
      lif_step_bf16_kernel<<<(int)(nBH / 256), 256, 0, stream>>>(
          delta1, mask_lif, mem_lif, h_bf, thr, leak);
      dim3 grid(H_SZ / BN, B_SZ / BM);
      gemm_async_bf16_wmma_kernel<<<grid, 256, 0, stream>>>(
          h_bf, W2bf, b2, mem_int, B_SZ, H_SZ, H_SZ, 1);
    }
    cvt_f32_bf16_kernel<<<(int)(nBH / 1024), 256, 0, stream>>>(mem_int, mi_bf);
    {
      dim3 grid((V_SZ + BN - 1) / BN, B_SZ / BM);
      gemm_async_bf16_wmma_kernel<<<grid, 256, 0, stream>>>(
          mi_bf, fcWbf, fcb, linear, B_SZ, V_SZ, H_SZ, 0);
    }
  } else {
    // ---- fallback: f32 sources, on-the-fly conversion (14 MB ws) ----
    float* ws      = (float*)d_ws;
    float* x       = ws;
    float* delta1  = x + nBE;
    float* mem_lif = delta1 + nBH;
    float* h       = mem_lif + nBH;

    embed_mask_f32_kernel<<<(int)(nBE / 256), 256, 0, stream>>>(tok, emb, mask_emb, x);
    init_state_kernel<<<(int)(nBH / 256), 256, 0, stream>>>(hidden, mem_lif, mem_int);
    {
      dim3 grid(H_SZ / BN, B_SZ / BM);
      gemm_bf16_wmma_kernel<<<grid, 256, 0, stream>>>(x, W1, b1, delta1,
                                                      B_SZ, H_SZ, E_SZ, 0);
    }
    for (int t = 0; t < TSTEPS; ++t) {
      lif_step_f32_kernel<<<(int)(nBH / 256), 256, 0, stream>>>(
          delta1, mask_lif, mem_lif, h, thr, leak);
      dim3 grid(H_SZ / BN, B_SZ / BM);
      gemm_bf16_wmma_kernel<<<grid, 256, 0, stream>>>(h, W2, b2, mem_int,
                                                      B_SZ, H_SZ, H_SZ, 1);
    }
    {
      dim3 grid((V_SZ + BN - 1) / BN, B_SZ / BM);
      gemm_bf16_wmma_kernel<<<grid, 256, 0, stream>>>(mem_int, fcW, fcb, linear,
                                                      B_SZ, V_SZ, H_SZ, 0);
    }
  }
}
